// MSARowAttentionWithPairBias_26637387170078
// MI455X (gfx1250) — matrix operand order repaired and checked by reading, not compiled, not run on hardware
//
#include <hip/hip_runtime.h>
#include <hip/hip_fp16.h>
#include <math.h>

typedef __attribute__((ext_vector_type(16))) _Float16 v16h;
typedef __attribute__((ext_vector_type(8)))  _Float16 v8h;
typedef __attribute__((ext_vector_type(8)))  float    v8f;
typedef __attribute__((ext_vector_type(4)))  float    v4f;

#define WMMA_F16(A, B, C) \
  __builtin_amdgcn_wmma_f32_16x16x32_f16(false, (A), false, (B), (short)0, (C), false, false)

namespace {
constexpr int kH = 8, kL = 256, kD = 256, kDH = 32, kN = 128, kP = 128;
constexpr int kRows = kN * kL;   // 32768
constexpr int k3D = 3 * kD;      // 768
}

// ---------------------------------------------------------------------------
// Kernel 1: convert + transpose weights fp32 -> f16 (B-operand wants W^T rows)
// ---------------------------------------------------------------------------
__global__ void cvt_weights_kernel(const float* __restrict__ wqkv,
                                   const float* __restrict__ wout,
                                   _Float16* __restrict__ wqkv_t,
                                   _Float16* __restrict__ wout_t) {
  int idx = blockIdx.x * blockDim.x + threadIdx.x;
  if (idx < k3D * kD) {
    int col = idx / kD, k = idx % kD;
    wqkv_t[(size_t)col * kD + k] = (_Float16)wqkv[(size_t)k * k3D + col];
  } else if (idx < k3D * kD + kD * kD) {
    int i2 = idx - k3D * kD;
    int col = i2 / kD, k = i2 % kD;
    wout_t[(size_t)col * kD + k] = (_Float16)wout[(size_t)k * kD + col];
  }
}

// ---------------------------------------------------------------------------
// Kernel 2: pair_bias[h][i][j] = sum_p pair[i][j][p] * w_pb[p][h] + b_pb[h]
// ---------------------------------------------------------------------------
__global__ void pair_bias_kernel(const float* __restrict__ pair,
                                 const float* __restrict__ wpb,
                                 const float* __restrict__ bpb,
                                 float* __restrict__ bias) {
  int ij = blockIdx.x * blockDim.x + threadIdx.x;
  if (ij >= kL * kL) return;
  const float* prow = pair + (size_t)ij * kP;
  float acc[kH];
#pragma unroll
  for (int h = 0; h < kH; ++h) acc[h] = bpb[h];
  for (int p = 0; p < kP; ++p) {
    float v = prow[p];
#pragma unroll
    for (int h = 0; h < kH; ++h) acc[h] += v * wpb[p * kH + h];
  }
#pragma unroll
  for (int h = 0; h < kH; ++h) bias[(size_t)h * kL * kL + ij] = acc[h];
}

// ---------------------------------------------------------------------------
// Kernel 3: QKV projection (32768x256 f32) x (256x768 f16^T) via WMMA.
//  A tile per lane: rows M = lane&15; k = 16*(lane>>4)+e (contiguous, as B)
//  B tile per lane: col N = lane&15, k = 16*(lane>>4)+e.
//  Writes Q/K as [n*H+h][l][dh] f16 and V transposed [n*H+h][dh][l] f16.
// ---------------------------------------------------------------------------
__global__ __launch_bounds__(128) void qkv_gemm_kernel(
    const float* __restrict__ msa, const _Float16* __restrict__ wqkv_t,
    _Float16* __restrict__ Qh, _Float16* __restrict__ Kh,
    _Float16* __restrict__ Vt) {
  const int lane = threadIdx.x & 31, wave = threadIdx.x >> 5;
  const int lm = lane & 15, hs = lane >> 4;
  const int rowTile = blockIdx.x * 64 + wave * 16;
  const int colBase = blockIdx.y * 64;
  const float* arow = msa + (size_t)(rowTile + lm) * kD;
  v8f acc[4] = {};
  for (int k0 = 0; k0 < kD; k0 += 32) {
    v16h a;
    const int o = k0 + hs * 16;
    v4f a0 = *(const v4f*)(arow + o);
    v4f a1 = *(const v4f*)(arow + o + 4);
    v4f a2 = *(const v4f*)(arow + o + 8);
    v4f a3 = *(const v4f*)(arow + o + 12);
#pragma unroll
    for (int e = 0; e < 4; ++e) {
      a[e] = (_Float16)a0[e];      a[4 + e]  = (_Float16)a1[e];
      a[8 + e] = (_Float16)a2[e];  a[12 + e] = (_Float16)a3[e];
    }
#pragma unroll
    for (int ct = 0; ct < 4; ++ct) {
      int col = colBase + ct * 16 + lm;
      v16h b = *(const v16h*)(wqkv_t + (size_t)col * kD + k0 + hs * 16);
      acc[ct] = WMMA_F16(a, b, acc[ct]);
    }
  }
#pragma unroll
  for (int ct = 0; ct < 4; ++ct) {
    int col = colBase + ct * 16 + lm;
    int s = col >> 8, rem = col & 255;   // s: 0=Q,1=K,2=V
    int h = rem >> 5, dh = rem & 31;
#pragma unroll
    for (int r = 0; r < 8; ++r) {
      int row = rowTile + r + 8 * hs;    // C layout: M = r + 8*(lane>>4)
      int n = row >> 8, li = row & 255;
      _Float16 v = (_Float16)acc[ct][r];
      size_t nh = (size_t)(n * kH + h);
      if (s == 0)      Qh[(nh * kL + li) * kDH + dh] = v;
      else if (s == 1) Kh[(nh * kL + li) * kDH + dh] = v;
      else             Vt[(nh * kDH + dh) * kL + li] = v;
    }
  }
}

// ---------------------------------------------------------------------------
// Kernel 4: attention per (n,h). 4 waves, each owns a 16-row q strip.
// S = Q*K^T (16 WMMAs, K=32=DH), bias+mask+softmax in regs (shfl_xor row
// reductions across 16-lane halves), P staged via LDS (C->A layout), O = P*V.
// ---------------------------------------------------------------------------
__global__ __launch_bounds__(128) void attn_kernel(
    const _Float16* __restrict__ Qh, const _Float16* __restrict__ Kh,
    const _Float16* __restrict__ Vt, const float* __restrict__ bias,
    const unsigned char* __restrict__ mask, _Float16* __restrict__ AO) {
  __shared__ __align__(32) _Float16 plds[4][16 * kL];   // 8 KB per wave
  const int nh = blockIdx.x;
  const int n = nh >> 3, h = nh & 7;
  const int lane = threadIdx.x & 31, wave = threadIdx.x >> 5;
  const int lm = lane & 15, hs = lane >> 4;
  const int off = hs * 16;
  const _Float16* Qb = Qh + (size_t)nh * kL * kDH;
  const _Float16* Kb = Kh + (size_t)nh * kL * kDH;
  const _Float16* Vb = Vt + (size_t)nh * kDH * kL;
  const float* bh = bias + (size_t)h * kL * kL;
  const unsigned char* mk = mask + n * kL;   // jax bool: 1 byte/elem
  const float scaling = 0.17677669529663687f;  // 32^-0.5
  _Float16* pl = &plds[wave][0];

  for (int qt = wave; qt < 16; qt += 4) {
    const int qbase = qt * 16;
    // ---- A = Q tile (16x32)
    v16h a;
    {
      const _Float16* qr = Qb + (size_t)(qbase + lm) * kDH;
      v8h q0 = *(const v8h*)(qr + off);
      v8h q1 = *(const v8h*)(qr + 8 + off);
#pragma unroll
      for (int e = 0; e < 8; ++e) { a[e] = q0[e]; a[8 + e] = q1[e]; }
    }
    // ---- S = Q*K^T : full 16x256 score block in registers
    v8f acc[16];
#pragma unroll
    for (int j = 0; j < 16; ++j) {
      v16h b = *(const v16h*)(Kb + (size_t)(j * 16 + lm) * kDH + hs * 16);
      v8f z = {};
      acc[j] = WMMA_F16(a, b, z);
    }
    // ---- scale + pair bias + mask; track row max
    float mx[8], sm[8];
#pragma unroll
    for (int r = 0; r < 8; ++r) mx[r] = -INFINITY;
#pragma unroll
    for (int j = 0; j < 16; ++j) {
      int col = j * 16 + lm;
      bool valid = mk[col] != 0;
      const float* bp = bh + (size_t)(qbase + 8 * hs) * kL + col;
#pragma unroll
      for (int r = 0; r < 8; ++r) {
        float s = acc[j][r] * scaling + bp[(size_t)r * kL];
        s = valid ? s : -INFINITY;
        acc[j][r] = s;
        mx[r] = fmaxf(mx[r], s);
      }
    }
#pragma unroll
    for (int r = 0; r < 8; ++r) {
#pragma unroll
      for (int m = 1; m < 16; m <<= 1) mx[r] = fmaxf(mx[r], __shfl_xor(mx[r], m, 32));
      sm[r] = 0.f;
    }
#pragma unroll
    for (int j = 0; j < 16; ++j)
#pragma unroll
      for (int r = 0; r < 8; ++r) {
        float p = __expf(acc[j][r] - mx[r]);
        acc[j][r] = p;
        sm[r] += p;
      }
#pragma unroll
    for (int r = 0; r < 8; ++r) {
#pragma unroll
      for (int m = 1; m < 16; m <<= 1) sm[r] += __shfl_xor(sm[r], m, 32);
      sm[r] = 1.f / sm[r];
    }
    // ---- stage normalized P (f16) in LDS, C layout -> row-major 16x256
#pragma unroll
    for (int j = 0; j < 16; ++j)
#pragma unroll
      for (int r = 0; r < 8; ++r)
        pl[(r + 8 * hs) * kL + j * 16 + lm] = (_Float16)(acc[j][r] * sm[r]);
    __syncthreads();
    // ---- O = P(16x256) x V(256x32): 2 N-tiles x 8 K-steps
#pragma unroll
    for (int t = 0; t < 2; ++t) {
      v8f o = {};
#pragma unroll
      for (int kk = 0; kk < 8; ++kk) {
        v16h pa;
        const _Float16* pr = pl + lm * kL + kk * 32;
        v8h p0 = *(const v8h*)(pr + off);
        v8h p1 = *(const v8h*)(pr + 8 + off);
#pragma unroll
        for (int e = 0; e < 8; ++e) { pa[e] = p0[e]; pa[8 + e] = p1[e]; }
        v16h vb = *(const v16h*)(Vb + (size_t)(t * 16 + lm) * kL + kk * 32 + hs * 16);
        o = WMMA_F16(pa, vb, o);
      }
#pragma unroll
      for (int r = 0; r < 8; ++r) {
        int row = n * kL + qbase + r + 8 * hs;
        int col = h * kDH + t * 16 + lm;
        AO[(size_t)row * kD + col] = (_Float16)o[r];
      }
    }
    __syncthreads();
  }
}

// ---------------------------------------------------------------------------
// Kernel 5: Out = AO(32768x256 f16) x Wout^T + b_out, f32 result
// ---------------------------------------------------------------------------
__global__ __launch_bounds__(128) void out_gemm_kernel(
    const _Float16* __restrict__ AO, const _Float16* __restrict__ wout_t,
    const float* __restrict__ bout, float* __restrict__ out) {
  const int lane = threadIdx.x & 31, wave = threadIdx.x >> 5;
  const int lm = lane & 15, hs = lane >> 4;
  const int rowTile = blockIdx.x * 64 + wave * 16;
  const int colBase = blockIdx.y * 64;
  const _Float16* arow = AO + (size_t)(rowTile + lm) * kD;
  v8f acc[4] = {};
  for (int k0 = 0; k0 < kD; k0 += 32) {
    v16h a;
    v8h a0 = *(const v8h*)(arow + k0 + hs * 16);
    v8h a1 = *(const v8h*)(arow + k0 + 8 + hs * 16);
#pragma unroll
    for (int e = 0; e < 8; ++e) { a[e] = a0[e]; a[8 + e] = a1[e]; }
#pragma unroll
    for (int ct = 0; ct < 4; ++ct) {
      int col = colBase + ct * 16 + lm;
      v16h b = *(const v16h*)(wout_t + (size_t)col * kD + k0 + hs * 16);
      acc[ct] = WMMA_F16(a, b, acc[ct]);
    }
  }
#pragma unroll
  for (int ct = 0; ct < 4; ++ct) {
    int col = colBase + ct * 16 + lm;
    float bv = bout[col];
#pragma unroll
    for (int r = 0; r < 8; ++r) {
      int row = rowTile + r + 8 * hs;
      out[(size_t)row * kD + col] = acc[ct][r] + bv;
    }
  }
}

// ---------------------------------------------------------------------------
extern "C" void kernel_launch(void* const* d_in, const int* in_sizes, int n_in,
                              void* d_out, int out_size, void* d_ws, size_t ws_size,
                              hipStream_t stream) {
  (void)in_sizes; (void)n_in; (void)out_size; (void)ws_size;
  const float* msa  = (const float*)d_in[0];
  const float* pair = (const float*)d_in[1];
  const unsigned char* mask = (const unsigned char*)d_in[2];  // jax bool (1B)
  const float* w_qkv = (const float*)d_in[3];
  const float* w_pb  = (const float*)d_in[4];
  const float* b_pb  = (const float*)d_in[5];
  const float* w_out = (const float*)d_in[6];
  const float* b_out = (const float*)d_in[7];
  float* out = (float*)d_out;

  char* ws = (char*)d_ws;
  auto alloc = [&](size_t bytes) -> char* {
    char* p = ws;
    ws += (bytes + 255) & ~(size_t)255;
    return p;
  };
  _Float16* wqkv_t = (_Float16*)alloc((size_t)k3D * kD * 2);          // 384 KB
  _Float16* wout_t = (_Float16*)alloc((size_t)kD * kD * 2);           // 128 KB
  float*    bias   = (float*)   alloc((size_t)kH * kL * kL * 4);      //   2 MB
  _Float16* Qh     = (_Float16*)alloc((size_t)kN * kH * kL * kDH * 2);// 16 MB
  _Float16* Kh     = (_Float16*)alloc((size_t)kN * kH * kL * kDH * 2);// 16 MB
  _Float16* Vt     = (_Float16*)alloc((size_t)kN * kH * kL * kDH * 2);// 16 MB
  _Float16* AO     = (_Float16*)alloc((size_t)kRows * kD * 2);        // 16 MB

  cvt_weights_kernel<<<(k3D * kD + kD * kD + 255) / 256, 256, 0, stream>>>(
      w_qkv, w_out, wqkv_t, wout_t);
  pair_bias_kernel<<<(kL * kL + 127) / 128, 128, 0, stream>>>(
      pair, w_pb, b_pb, bias);
  qkv_gemm_kernel<<<dim3(kRows / 64, k3D / 64), 128, 0, stream>>>(
      msa, wqkv_t, Qh, Kh, Vt);
  attn_kernel<<<kN * kH, 128, 0, stream>>>(Qh, Kh, Vt, bias, mask, AO);
  out_gemm_kernel<<<dim3(kRows / 64, kD / 64), 128, 0, stream>>>(
      AO, wout_t, b_out, out);
}
